// VanillaRNN_68066641707050
// MI455X (gfx1250) — matrix operand order repaired and checked by reading, not compiled, not run on hardware
//
#include <hip/hip_runtime.h>
#include <hip/hip_bf16.h>
#include <math.h>

#define VOCAB 512
#define HID   2048
#define BATCH 256
#define SEQ   512

typedef __attribute__((ext_vector_type(8)))  float  v8f;
typedef __attribute__((ext_vector_type(16))) __bf16 v16bf;
typedef __attribute__((ext_vector_type(8)))  __bf16 v8bf;

#if __has_builtin(__builtin_amdgcn_sched_group_barrier)
// mask 0x20 = VMEM read group, 0x8 = MFMA/WMMA group
#define SCHED_VMEM(n)  __builtin_amdgcn_sched_group_barrier(0x020, (n), 0)
#define SCHED_WMMA(n)  __builtin_amdgcn_sched_group_barrier(0x008, (n), 0)
#else
#define SCHED_VMEM(n)
#define SCHED_WMMA(n)
#endif

// ---------------------------------------------------------------------------
// Fragment loaders. 32-bit element offsets from a uniform base pointer so the
// backend uses saddr + vgpr32 + imm24 addressing; K-loop fully unrolled so
// kt*stride folds into the immediate offset.
// ---------------------------------------------------------------------------
__device__ __forceinline__ v16bf load_a(const __bf16* __restrict__ p,
                                        unsigned base, int kt) {
    v8bf lo = *(const v8bf*)(p + base + kt * 32u);
    v8bf hi = *(const v8bf*)(p + base + kt * 32u + 8u);
    return __builtin_shufflevector(lo, hi,
             0,1,2,3,4,5,6,7,8,9,10,11,12,13,14,15);
}
__device__ __forceinline__ v16bf load_b(const __bf16* __restrict__ p,
                                        unsigned base, int kt) {
    return *(const v16bf*)(p + base + kt * 512u);   // 32 lanes * 16 elems
}
__device__ __forceinline__ v8f wmma_bf16(v16bf a, v16bf b, v8f c) {
    return __builtin_amdgcn_wmma_f32_16x16x32_bf16(
               false, a, false, b, (short)0, c, false, false);
}

// ---------------------------------------------------------------------------
// Shared 16x64-per-wave GEMM core with 2-deep register pipeline:
// fragments for kt+2 are issued before the WMMAs of kt; sched_group_barrier
// pins the VMEM-then-WMMA pattern so the allocator keeps stage registers
// distinct and every WMMA waits with ~20 loads still in flight.
// ---------------------------------------------------------------------------
__device__ __forceinline__ void gemm_16x64(const __bf16* __restrict__ A,
                                           const __bf16* __restrict__ B,
                                           unsigned aBase, const unsigned* bBase,
                                           v8f c[4]) {
    v16bf a0 = load_a(A, aBase, 0);
    v16bf a1 = load_a(A, aBase, 1);
    v16bf b0[4], b1[4];
#pragma unroll
    for (int j = 0; j < 4; ++j) b0[j] = load_b(B, bBase[j], 0);
#pragma unroll
    for (int j = 0; j < 4; ++j) b1[j] = load_b(B, bBase[j], 1);

#pragma unroll
    for (int kt = 0; kt < 64; ++kt) {
        v16bf a2 = a1;
        v16bf b2[4];
#pragma unroll
        for (int j = 0; j < 4; ++j) b2[j] = b1[j];
        if (kt + 2 < 64) {
            a2 = load_a(A, aBase, kt + 2);
#pragma unroll
            for (int j = 0; j < 4; ++j) b2[j] = load_b(B, bBase[j], kt + 2);
        }
#pragma unroll
        for (int j = 0; j < 4; ++j) c[j] = wmma_bf16(a0, b0[j], c[j]);
        if (kt + 2 < 64) SCHED_VMEM(10);   // 2x A b128 + 8x B b128
        SCHED_WMMA(4);
        a0 = a1; a1 = a2;
#pragma unroll
        for (int j = 0; j < 4; ++j) { b0[j] = b1[j]; b1[j] = b2[j]; }
    }
}

// ---------------------------------------------------------------------------
// Preamble: pack a K x N (K=2048) fp32 weight view into WMMA B-fragment order:
//   dst[((ntile*64 + ktile)*32 + lane)*16 + e] = src[n*rowStride + colOff + k]
//   n = ntile*16 + (lane&15);  k = ktile*32 + (lane>>4)*16 + e
// ---------------------------------------------------------------------------
__global__ __launch_bounds__(256)
void rnn_pack_b(const float* __restrict__ src, int rowStride, int colOff,
                __bf16* __restrict__ dst) {
    const int tid  = blockIdx.x * 256 + threadIdx.x;
    const int lane = tid & 31;
    const int rest = tid >> 5;          // ntile*64 + ktile
    const int ktile = rest & 63;        // KT = 2048/32 = 64
    const int ntile = rest >> 6;
    const int n  = ntile * 16 + (lane & 15);
    const int kb = ktile * 32 + (lane >> 4) * 16;
    const float* s = src + n * rowStride + colOff + kb;
    __bf16* d = dst + (size_t)tid * 16;
#pragma unroll
    for (int e = 0; e < 16; ++e) d[e] = (__bf16)s[e];
}

// WxT[v][h] = W_i2h[h][v]  (fp32, 512 x 2048) — coalesced writes
__global__ __launch_bounds__(256)
void rnn_pack_wxt(const float* __restrict__ Wi2h, float* __restrict__ WxT) {
    const int tid = blockIdx.x * 256 + threadIdx.x;   // over 512*2048
    const int v = tid >> 11;
    const int h = tid & 2047;
    WxT[tid] = Wi2h[h * (VOCAB + HID) + v];
}

__global__ __launch_bounds__(256)
void rnn_init_h(const float* __restrict__ h0, __bf16* __restrict__ Hb) {
    const int tid = blockIdx.x * 256 + threadIdx.x;   // over 256*2048
    Hb[tid] = (__bf16)h0[tid];
}

// ---------------------------------------------------------------------------
// Step kernel 1: Hnew = tanh(Hb_in @ Wh_T + WxT[tok] + b_i2h)
// grid = 64 blocks x 256 thr (8 waves); wave -> 16x64 output strip.
// ---------------------------------------------------------------------------
__global__ __launch_bounds__(256)
void rnn_step_h(const __bf16* __restrict__ Hb_in, const __bf16* __restrict__ WhP,
                const float* __restrict__ WxT, const float* __restrict__ b_i2h,
                const int* __restrict__ X, int t,
                __bf16* __restrict__ Hb_out, float* __restrict__ Hf_out) {
    const int lane   = threadIdx.x & 31;
    const int wave   = threadIdx.x >> 5;
    const int gw     = blockIdx.x * 8 + wave;   // 0..511
    const int mtile  = gw >> 5;                 // 0..15
    const int ncol   = gw & 31;                 // 0..31
    const int n0tile = ncol * 4;
    const int rowA   = mtile * 16 + (lane & 15);
    const int selA   = (lane >> 4) * 16;        // A frag K-half select
    const int mHi    = (lane >> 4) * 8;         // C frag M offset

    const unsigned aBase = (unsigned)(rowA * HID + selA);
    unsigned bBase[4];
#pragma unroll
    for (int j = 0; j < 4; ++j)
        bBase[j] = (unsigned)((((n0tile + j) * 64) * 32 + lane) * 16);

    v8f c[4] = {};
    gemm_16x64(Hb_in, WhP, aBase, bBase, c);

    int tok[8];
#pragma unroll
    for (int e = 0; e < 8; ++e) {
        const unsigned row = (unsigned)(mtile * 16 + mHi + e);
        tok[e] = X[row * SEQ + t];
    }
#pragma unroll
    for (int j = 0; j < 4; ++j) {
        const unsigned col = (unsigned)((n0tile + j) * 16 + (lane & 15));
        const float bi = b_i2h[col];
#pragma unroll
        for (int e = 0; e < 8; ++e) {
            const unsigned row = (unsigned)(mtile * 16 + mHi + e);
            const float v  = c[j][e] + WxT[(unsigned)tok[e] * HID + col] + bi;
            const float hv = tanhf(v);
            Hb_out[row * HID + col] = (__bf16)hv;
            if (Hf_out) Hf_out[row * HID + col] = hv;
        }
    }
}

// ---------------------------------------------------------------------------
// Step kernel 2: Y = Hb @ W_h2o_T + b_h2o  -> d_out + t*B*V (fp32)
// grid = 16 blocks x 256 thr; wave -> 16x64 strip of [256 x 512]
// ---------------------------------------------------------------------------
__global__ __launch_bounds__(256)
void rnn_step_y(const __bf16* __restrict__ Hb, const __bf16* __restrict__ WoP,
                const float* __restrict__ b_h2o, float* __restrict__ Yout) {
    const int lane   = threadIdx.x & 31;
    const int wave   = threadIdx.x >> 5;
    const int gw     = blockIdx.x * 8 + wave;   // 0..127
    const int mtile  = gw >> 3;                 // 0..15
    const int ncol   = gw & 7;                  // 0..7
    const int n0tile = ncol * 4;
    const int rowA   = mtile * 16 + (lane & 15);
    const int selA   = (lane >> 4) * 16;
    const int mHi    = (lane >> 4) * 8;

    const unsigned aBase = (unsigned)(rowA * HID + selA);
    unsigned bBase[4];
#pragma unroll
    for (int j = 0; j < 4; ++j)
        bBase[j] = (unsigned)((((n0tile + j) * 64) * 32 + lane) * 16);

    v8f c[4] = {};
    gemm_16x64(Hb, WoP, aBase, bBase, c);

#pragma unroll
    for (int j = 0; j < 4; ++j) {
        const unsigned col = (unsigned)((n0tile + j) * 16 + (lane & 15));
        const float bo = b_h2o[col];
#pragma unroll
        for (int e = 0; e < 8; ++e) {
            const unsigned row = (unsigned)(mtile * 16 + mHi + e);
            Yout[row * VOCAB + col] = c[j][e] + bo;
        }
    }
}

// ---------------------------------------------------------------------------
extern "C" void kernel_launch(void* const* d_in, const int* in_sizes, int n_in,
                              void* d_out, int out_size, void* d_ws, size_t ws_size,
                              hipStream_t stream) {
    const int*   X      = (const int*)  d_in[0];   // [B, S] int32
    const float* h0     = (const float*)d_in[1];   // [B, H]
    const float* W_i2h  = (const float*)d_in[2];   // [H, V+H]
    const float* b_i2h  = (const float*)d_in[3];   // [H]
    const float* W_h2o  = (const float*)d_in[4];   // [V, H]
    const float* b_h2o  = (const float*)d_in[5];   // [V]
    float* out = (float*)d_out;                    // [S*B*V] ++ [B*H]

    // workspace layout (~16 MB)
    char* ws = (char*)d_ws;
    __bf16* WhP = (__bf16*)(ws);                               // 8 MB
    __bf16* WoP = (__bf16*)(ws + (size_t)8  * 1024 * 1024);    // 2 MB
    float*  WxT = (float*) (ws + (size_t)10 * 1024 * 1024);    // 4 MB
    __bf16* Hb0 = (__bf16*)(ws + (size_t)14 * 1024 * 1024);    // 1 MB
    __bf16* Hb1 = (__bf16*)(ws + (size_t)15 * 1024 * 1024);    // 1 MB

    // one-time (per call) weight packing + state init
    rnn_pack_b<<<1024, 256, 0, stream>>>(W_i2h, VOCAB + HID, VOCAB, WhP); // Wh_T
    rnn_pack_b<<<256,  256, 0, stream>>>(W_h2o, HID, 0, WoP);            // W_h2o_T
    rnn_pack_wxt<<<4096, 256, 0, stream>>>(W_i2h, WxT);
    rnn_init_h<<<2048, 256, 0, stream>>>(h0, Hb0);

    float* hf_out = out + (size_t)SEQ * BATCH * VOCAB;  // final hidden state
    for (int t = 0; t < SEQ; ++t) {
        __bf16* hin  = (t & 1) ? Hb1 : Hb0;
        __bf16* hout = (t & 1) ? Hb0 : Hb1;
        rnn_step_h<<<64, 256, 0, stream>>>(hin, WhP, WxT, b_i2h, X, t, hout,
                                           (t == SEQ - 1) ? hf_out : nullptr);
        rnn_step_y<<<16, 256, 0, stream>>>(hout, WoP, b_h2o,
                                           out + (size_t)t * BATCH * VOCAB);
    }
}
